// QuantLinear_6210522710593
// MI455X (gfx1250) — matrix operand order repaired and checked by reading, not compiled, not run on hardware
//
#include <hip/hip_runtime.h>

typedef __attribute__((ext_vector_type(16))) _Float16 v16h;
typedef __attribute__((ext_vector_type(8)))  _Float16 v8h;
typedef __attribute__((ext_vector_type(4)))  _Float16 v4h;
typedef __attribute__((ext_vector_type(8)))  float    v8f;
typedef __attribute__((ext_vector_type(4)))  float    v4f;
typedef __attribute__((ext_vector_type(4)))  int      v4i;

// Problem dims (reference: M,N,K = 4096, group size 128)
constexpr int Mdim = 4096;
constexpr int Ndim = 4096;
constexpr int Kdim = 4096;
constexpr int Grp  = 128;
constexpr int NGrp = Kdim / Grp;      // 32 groups per weight row

// Tiling: 256 threads = 8 wave32 (4 in M x 2 in N), each wave owns 64x64.
constexpr int BM  = 256;
constexpr int BN  = 128;
constexpr int BK  = 32;
constexpr int LDT = BK + 8;           // padded LDS row stride (halves), kills bank conflicts
constexpr int NKT = Kdim / BK;        // 128 K-tiles; BK divides Grp -> group = kt/4

#if __has_builtin(__builtin_amdgcn_global_load_async_to_lds_b128)
#define ASYNC_A 1
typedef __attribute__((address_space(1))) v4i gv4i;   // global-AS int4 (builtin's param type)
typedef __attribute__((address_space(3))) v4i lv4i;   // LDS-AS int4
#else
#define ASYNC_A 0
#endif

static __device__ __forceinline__ void wait_async0() {
#if __has_builtin(__builtin_amdgcn_s_wait_asynccnt)
    __builtin_amdgcn_s_wait_asynccnt(0);
#else
    asm volatile("s_wait_asynccnt 0" ::: "memory");
#endif
}

// ---------------- pre-pass: x f32 -> f16 (done ONCE, not 32x per N-block) ---------
__global__ __launch_bounds__(256)
void x_to_f16(const float* __restrict__ x, _Float16* __restrict__ xh)
{
    const size_t i = ((size_t)blockIdx.x * 256 + threadIdx.x) * 8;
    v4f a = *(const v4f*)(x + i);
    v4f b = *(const v4f*)(x + i + 4);
    v8h h;
    #pragma unroll
    for (int j = 0; j < 4; ++j) { h[j] = (_Float16)a[j]; h[j + 4] = (_Float16)b[j]; }
    *(v8h*)(xh + i) = h;
}

// ---------------- main fused dequant-GEMM -----------------------------------------
__global__ __launch_bounds__(256)
void qlinear_wmma_f16(const _Float16* __restrict__ xh,
                      const int*      __restrict__ qw,
                      const float*    __restrict__ scales,
                      const float*    __restrict__ zeros,
                      const float*    __restrict__ bias,
                      float*          __restrict__ out)
{
    __shared__ _Float16 As[2][BM * LDT];   // x tile, f16           (20 KB / buf)
    __shared__ _Float16 Bs[2][BN * LDT];   // dequant W tile, f16   (10 KB / buf)

    const int tid  = threadIdx.x;
    const int lane = tid & 31;
    const int wid  = tid >> 5;
    const int wm   = (wid & 3) * 64;       // wave M offset inside block tile
    const int wn   = (wid >> 2) * 64;      // wave N offset inside block tile
    const int m0   = blockIdx.y * BM;
    const int n0   = blockIdx.x * BN;

    // A-tile copy mapping: 256 rows x 64B; thread t covers rows (t/4)+64i, chunk t%4
    const int arow = tid >> 2;             // 0..63
    const int acol = (tid & 3) * 8;        // halves: 0,8,16,24 (16B chunks)
#if !ASYNC_A
    v8h aregh[4];                          // fallback staging (f16 pass-through)
#endif

    // B-tile: 128 rows x 32 i32; thread covers rows frow+32i, k-offset fcol
    const int frow = tid >> 3;             // 0..31
    const int fcol = (tid & 7) * 4;        // 0,4,...,28
    v4i   breg[4];
    float sreg[4], zreg[4];

    auto issue_a = [&](int kt, int buf) {
        #pragma unroll
        for (int i = 0; i < 4; ++i) {
            const int r = arow + 64 * i;
            const _Float16* gp = xh + (size_t)(m0 + r) * Kdim + kt * BK + acol;
            _Float16*       lp = &As[buf][r * LDT + acol];
#if ASYNC_A
            __builtin_amdgcn_global_load_async_to_lds_b128((gv4i*)gp, (lv4i*)lp, 0, 0);
#else
            aregh[i] = *(const v8h*)gp;
            (void)lp;
#endif
        }
    };
#if !ASYNC_A
    auto commit_a = [&](int buf) {
        #pragma unroll
        for (int i = 0; i < 4; ++i)
            *(v8h*)(&As[buf][(arow + 64 * i) * LDT + acol]) = aregh[i];
    };
#endif

    auto load_b = [&](int kt) {
        const unsigned kbase = (unsigned)kt * BK + fcol;
        #pragma unroll
        for (int i = 0; i < 4; ++i) {
            const unsigned r = frow + 32 * i;
            breg[i] = *(const v4i*)(qw + (size_t)(n0 + r) * Kdim + kbase);
        }
        if ((kt & 3) == 0) {               // quant group changes every 4 K-tiles
            const unsigned g = (unsigned)kt >> 2;
            #pragma unroll
            for (int i = 0; i < 4; ++i) {
                const unsigned r = n0 + frow + 32 * i;
                sreg[i] = scales[r * NGrp + g];
                zreg[i] = zeros [r * NGrp + g];
            }
        }
    };

    auto store_b = [&](int buf) {
        #pragma unroll
        for (int i = 0; i < 4; ++i) {
            // asymmetric int4 dequant: w = q*s + (-z*s), packed f16 pipeline:
            // v_cvt_f32_i32 x4 -> v_cvt_pk_f16_f32 x2 -> v_pk_fma_f16 x2
            v4f qf;
            #pragma unroll
            for (int j = 0; j < 4; ++j) qf[j] = (float)breg[i][j];
            v4h qh;
            #pragma unroll
            for (int j = 0; j < 4; ++j) qh[j] = (_Float16)qf[j];
            const _Float16 sh = (_Float16)sreg[i];
            const _Float16 ch = (_Float16)(-zreg[i] * sreg[i]);
            const v4h sh4 = {sh, sh, sh, sh};
            const v4h ch4 = {ch, ch, ch, ch};
            v4h bh = qh * sh4 + ch4;
            *(v4h*)(&Bs[buf][(frow + 32 * i) * LDT + fcol]) = bh;
        }
    };

    v8f acc[4][4] = {};   // 4 (M) x 4 (N) 16x16 f32 accumulators per wave

    auto compute = [&](int buf) {
        // B fragments (32x16): lanes<16 hold K 0..15, lanes>=16 K 16..31; col = lane%16
        v16h bf[4];
        {
            const int kb = (lane >> 4) * 16;
            #pragma unroll
            for (int ni = 0; ni < 4; ++ni) {
                const int n = wn + ni * 16 + (lane & 15);
                v8h lo = *(const v8h*)(&Bs[buf][n * LDT + kb]);
                v8h hi = *(const v8h*)(&Bs[buf][n * LDT + kb + 8]);
                bf[ni] = __builtin_shufflevector(lo, hi,
                          0,1,2,3,4,5,6,7,8,9,10,11,12,13,14,15);
            }
        }
        // A fragments (16x32): lanes<16 hold K 0..15, lanes>=16 K 16..31; row = lane%16
        {
            const int kb = (lane >> 4) * 16;
            #pragma unroll
            for (int mi = 0; mi < 4; ++mi) {
                const int m = wm + mi * 16 + (lane & 15);
                v8h lo = *(const v8h*)(&As[buf][m * LDT + kb]);
                v8h hi = *(const v8h*)(&As[buf][m * LDT + kb + 8]);
                v16h af = __builtin_shufflevector(lo, hi,
                           0,1,2,3,4,5,6,7,8,9,10,11,12,13,14,15);
                #pragma unroll
                for (int ni = 0; ni < 4; ++ni)
                    acc[mi][ni] = __builtin_amdgcn_wmma_f32_16x16x32_f16(
                        false, af, false, bf[ni],
                        (short)0, acc[mi][ni], false, false);
            }
        }
    };

    // --- pipeline prologue ---
    issue_a(0, 0);                 // async: direct to LDS buf0 ; fallback: -> regs
    load_b(0);
#if !ASYNC_A
    commit_a(0);
#endif
    store_b(0);
#if !ASYNC_A
    issue_a(1, 1);                 // fallback: stage A regs for tile 1
#endif
    load_b(1);
#if ASYNC_A
    wait_async0();
#endif
    __syncthreads();

    // --- main loop: double-buffered LDS, 1 barrier/iter ---
    for (int kt = 0; kt < NKT; ++kt) {
        const int cur = kt & 1;
#if ASYNC_A
        if (kt + 1 < NKT) issue_a(kt + 1, cur ^ 1);  // async copy overlaps compute
#endif
        compute(cur);
        if (kt + 1 < NKT) {
#if !ASYNC_A
            commit_a(cur ^ 1);
#endif
            store_b(cur ^ 1);
            if (kt + 2 < NKT) {
#if !ASYNC_A
                issue_a(kt + 2, cur);   // fallback: restage A regs
#endif
                load_b(kt + 2);
            }
        }
#if ASYNC_A
        if (kt + 1 < NKT) wait_async0();
#endif
        __syncthreads();
    }

    // --- epilogue: bias add + coalesced f32 stores ---
    // C/D layout: VGPR r holds (row = r + 8*(lane/16), col = lane%16)
    #pragma unroll
    for (int ni = 0; ni < 4; ++ni) {
        const int n    = n0 + wn + ni * 16 + (lane & 15);
        const float bn = bias[n];
        #pragma unroll
        for (int mi = 0; mi < 4; ++mi) {
            #pragma unroll
            for (int r = 0; r < 8; ++r) {
                const int m = m0 + wm + mi * 16 + r + 8 * (lane >> 4);
                out[(size_t)m * Ndim + n] = acc[mi][ni][r] + bn;
            }
        }
    }
}

extern "C" void kernel_launch(void* const* d_in, const int* in_sizes, int n_in,
                              void* d_out, int out_size, void* d_ws, size_t ws_size,
                              hipStream_t stream)
{
    const float* x      = (const float*)d_in[0];
    const int*   qw     = (const int*)  d_in[1];
    const float* scales = (const float*)d_in[2];
    const float* zeros  = (const float*)d_in[3];
    const float* bias   = (const float*)d_in[4];
    float*       out    = (float*)d_out;
    _Float16*    xh     = (_Float16*)d_ws;          // 32 MB f16 staging of x

    // pre-pass: convert x once (4096*4096 / (256*8) = 8192 blocks)
    x_to_f16<<<(Mdim * Kdim) / (256 * 8), 256, 0, stream>>>(x, xh);

    dim3 grid(Ndim / BN, Mdim / BM);   // 32 x 16 blocks
    qlinear_wmma_f16<<<grid, 256, 0, stream>>>(xh, qw, scales, zeros, bias, out);
}
